// GNN_Model_83150566850820
// MI455X (gfx1250) — matrix operand order repaired and checked by reading, not compiled, not run on hardware
//
#include <hip/hip_runtime.h>
#include <hip/hip_bf16.h>

typedef _Float16 v16h __attribute__((ext_vector_type(16)));
typedef _Float16 v8h  __attribute__((ext_vector_type(8)));
typedef float    v8f  __attribute__((ext_vector_type(8)));

#define NEG_SLOPE 0.2f

static inline int cdiv(int a, int b) { return (a + b - 1) / b; }

// ---------------------------------------------------------------- utilities

__global__ void fill_f32_kernel(float* __restrict__ p, float v, int n) {
  int i = blockIdx.x * blockDim.x + threadIdx.x;
  if (i < n) p[i] = v;
}

__global__ void cvt_f16_kernel(const float* __restrict__ src,
                               _Float16* __restrict__ dst, int n) {
  int i = blockIdx.x * blockDim.x + threadIdx.x;
  if (i < n) dst[i] = (_Float16)src[i];
}

// Pack W[K,N] (f32) into WMMA B-fragment order (f16):
// tile (kt,nt) of shape 32(K)x16(N) -> 512 contiguous halves:
//   [lane 0..31][j 0..15] = W[kt*32 + (j>>3)*16 + (lane>>4)*8 + (j&7)][nt*16 + (lane&15)]
// so in the GEMM each lane reads its 16 halves as two aligned b128 loads.
__global__ void pack_w_kernel(const float* __restrict__ W,
                              _Float16* __restrict__ Wp, int K, int N) {
  int idx = blockIdx.x * blockDim.x + threadIdx.x;
  if (idx >= K * N) return;
  int j    = idx & 15;
  int lane = (idx >> 4) & 31;
  int tile = idx >> 9;                 // kt*(N/16) + nt
  int ntn  = N >> 4;
  int kt   = tile / ntn;
  int nt   = tile - kt * ntn;
  int k    = kt * 32 + (j >> 3) * 16 + (lane >> 4) * 8 + (j & 7);
  int col  = nt * 16 + (lane & 15);
  Wp[idx] = (_Float16)W[k * N + col];
}

// ---------------------------------------------------------------- GEMM (WMMA)
// C[M,N] = A[M,K] (f16, row-major) x Bp (f16, fragment-packed), f32 accum.
// One wave computes one 16x16 tile via v_wmma_f32_16x16x32_f16.
// Inner loop: 2x b128 (A) + 2x b128 (B) + 1x WMMA.
__global__ void gemm_f16_wmma_kernel(const _Float16* __restrict__ A,
                                     const _Float16* __restrict__ Bp,
                                     float* __restrict__ C,
                                     int M, int K, int N) {
  const int lane = threadIdx.x & 31;
  const int wave = threadIdx.x >> 5;
  const int m0 = blockIdx.y << 4;
  const int n0 = (blockIdx.x << 6) + (wave << 4);
  if (n0 >= N) return;
  const int grp  = lane >> 4;   // 0 or 1
  const int lrow = lane & 15;

  int arow = m0 + lrow;
  if (arow >= M) arow = M - 1;                 // clamp (M is 16-divisible here)
  const _Float16* __restrict__ Ap = A + arow * K + grp * 8;
  // packed B: start at tile (kt=0, nt=n0/16), this lane's 16 halves
  const _Float16* __restrict__ Bq = Bp + ((n0 >> 4) << 9) + (lane << 4);
  const int bstride = (N >> 4) << 9;           // halves to next kt

  v8f acc = {};
#pragma unroll 2
  for (int kb = 0; kb < K; kb += 32) {
    union { v16h v; v8h h[2]; } a, b;
    a.h[0] = *(const v8h*)(Ap + kb);           // K = grp*8 .. +7
    a.h[1] = *(const v8h*)(Ap + kb + 16);      // K = 16+grp*8 .. +7
    b.h[0] = *(const v8h*)(Bq);
    b.h[1] = *(const v8h*)(Bq + 8);
    Bq += bstride;
    acc = __builtin_amdgcn_wmma_f32_16x16x32_f16(false, a.v, false, b.v,
                                                 (short)0, acc, false, false);
  }
  // D layout: VGPR i -> M = grp*8 + i, N = lane&15
  const int bcol = n0 + lrow;
#pragma unroll
  for (int i = 0; i < 8; ++i) {
    int row = m0 + grp * 8 + i;
    if (row < M) C[row * N + bcol] = acc[i];
  }
}

// ---------------------------------------------------------------- attention

// s[n,h] = dot(xp[n,h,:], a_src[h,:]);  d[n,h] = dot(xp[n,h,:], a_dst[h,:])
__global__ void sd_kernel(const float* __restrict__ xp,
                          const float* __restrict__ a_src,
                          const float* __restrict__ a_dst,
                          float* __restrict__ s, float* __restrict__ d,
                          int N, int H) {
  int i = blockIdx.x * blockDim.x + threadIdx.x;
  if (i >= N * H) return;
  int n = i / H, h = i % H;
  const float* row = xp + (size_t)n * H * 64 + (size_t)h * 64;
  const float* as = a_src + h * 64;
  const float* ad = a_dst + h * 64;
  float ss = 0.f, dd = 0.f;
#pragma unroll 16
  for (int c = 0; c < 64; ++c) { float v = row[c]; ss += v * as[c]; dd += v * ad[c]; }
  s[i] = ss; d[i] = dd;
}

__device__ __forceinline__ void edge_nodes(const int* __restrict__ ei, int E,
                                           int eid, int& src, int& dst) {
  if (eid < E) { src = ei[eid]; dst = ei[E + eid]; }
  else         { src = eid - E; dst = eid - E; }        // self loops
}

__device__ __forceinline__ float lrelu(float x) {
  return x > 0.f ? x : NEG_SLOPE * x;
}

// monotonic int/uint mapping trick -> native global atomics
__device__ __forceinline__ void atomicMaxF(float* addr, float v) {
  if (v >= 0.f) atomicMax((int*)addr, __float_as_int(v));
  else          atomicMin((unsigned int*)addr, __float_as_uint(v));
}

__global__ void edge_max_kernel(const int* __restrict__ ei, int E, int ET,
                                const float* __restrict__ s,
                                const float* __restrict__ d,
                                float* __restrict__ emax, int H) {
  int idx = blockIdx.x * blockDim.x + threadIdx.x;
  if (idx >= ET * H) return;
  int eid = idx / H, h = idx - eid * H;
  int src, dst; edge_nodes(ei, E, eid, src, dst);
  float e = lrelu(s[src * H + h] + d[dst * H + h]);
  atomicMaxF(&emax[dst * H + h], e);
}

// ee = exp(e - emax[dst]); store ee into alpha buf; denom[dst] += ee
__global__ void edge_ee_kernel(const int* __restrict__ ei, int E, int ET,
                               const float* __restrict__ s,
                               const float* __restrict__ d,
                               const float* __restrict__ emax,
                               float* __restrict__ alpha,
                               float* __restrict__ denom, int H) {
  int idx = blockIdx.x * blockDim.x + threadIdx.x;
  if (idx >= ET * H) return;
  int eid = idx / H, h = idx - eid * H;
  int src, dst; edge_nodes(ei, E, eid, src, dst);
  float e = lrelu(s[src * H + h] + d[dst * H + h]);
  float m = emax[dst * H + h];
  if (!isfinite(m)) m = 0.f;
  float ee = __expf(e - m);
  alpha[idx] = ee;
  atomicAdd(&denom[dst * H + h], ee);
}

__global__ void alpha_div_kernel(const int* __restrict__ ei, int E, int ET,
                                 float* __restrict__ alpha,
                                 const float* __restrict__ denom, int H) {
  int idx = blockIdx.x * blockDim.x + threadIdx.x;
  if (idx >= ET * H) return;
  int eid = idx / H, h = idx - eid * H;
  int src, dst; edge_nodes(ei, E, eid, src, dst);
  alpha[idx] = alpha[idx] / (denom[dst * H + h] + 1e-16f);
}

// agg[dst, c..c+3] += alpha[e, c>>6] * xp[src, c..c+3]   (float4 per thread)
__global__ void edge_agg_kernel(const int* __restrict__ ei, int E, int ET,
                                const float* __restrict__ alpha,
                                const float* __restrict__ xp,
                                float* __restrict__ agg, int H, int CH) {
  int tpe = CH >> 2;                         // threads per edge (float4 each)
  int idx = blockIdx.x * blockDim.x + threadIdx.x;
  int eid = idx / tpe;
  int t   = idx - eid * tpe;
  if (eid >= ET) return;
  int src, dst; edge_nodes(ei, E, eid, src, dst);
  int c = t << 2;
  int h = c >> 6;
  float al = alpha[(size_t)eid * H + h];
  const float4 v = *(const float4*)(xp + (size_t)src * CH + c);
  float* o = agg + (size_t)dst * CH + c;
  atomicAdd(o + 0, al * v.x);
  atomicAdd(o + 1, al * v.y);
  atomicAdd(o + 2, al * v.z);
  atomicAdd(o + 3, al * v.w);
}

__global__ void bias_relu_kernel(float* __restrict__ h,
                                 const float* __restrict__ b, int n, int CH) {
  int i = blockIdx.x * blockDim.x + threadIdx.x;
  if (i >= n) return;
  float v = h[i] + b[i & (CH - 1)];          // CH is a power of two (256/64)
  h[i] = v > 0.f ? v : 0.f;
}

// out[n, 0..9] = h[n,:] @ Wfc + bfc
__global__ void fc_kernel(const float* __restrict__ h,
                          const float* __restrict__ W,
                          const float* __restrict__ b,
                          float* __restrict__ out, int N) {
  int n = blockIdx.x * blockDim.x + threadIdx.x;
  if (n >= N) return;
  float acc[10];
#pragma unroll
  for (int j = 0; j < 10; ++j) acc[j] = b[j];
  const float* row = h + (size_t)n * 64;
#pragma unroll 4
  for (int c = 0; c < 64; ++c) {
    float v = row[c];
#pragma unroll
    for (int j = 0; j < 10; ++j) acc[j] += v * W[c * 10 + j];
  }
#pragma unroll
  for (int j = 0; j < 10; ++j) out[(size_t)n * 10 + j] = acc[j];
}

// ---------------------------------------------------------------- driver

extern "C" void kernel_launch(void* const* d_in, const int* in_sizes, int n_in,
                              void* d_out, int out_size, void* d_ws, size_t ws_size,
                              hipStream_t stream) {
  const float* x   = (const float*)d_in[0];
  const int*   ei  = (const int*)d_in[1];
  const float* Wl[3]    = { (const float*)d_in[2],  (const float*)d_in[6],  (const float*)d_in[10] };
  const float* asrc[3]  = { (const float*)d_in[3],  (const float*)d_in[7],  (const float*)d_in[11] };
  const float* adst[3]  = { (const float*)d_in[4],  (const float*)d_in[8],  (const float*)d_in[12] };
  const float* bl[3]    = { (const float*)d_in[5],  (const float*)d_in[9],  (const float*)d_in[13] };
  const float* Wfc = (const float*)d_in[14];
  const float* bfc = (const float*)d_in[15];
  float* out = (float*)d_out;

  const int N  = in_sizes[0] / 128;
  const int E  = in_sizes[1] / 2;
  const int ET = E + N;                               // + self loops
  const int Cin[3]  = { 128, 256, 256 };
  const int Cout[3] = { 256, 256, 64  };
  const int Hh[3]   = { 4, 4, 1 };

  // ---- carve workspace
  size_t off = 0;
  auto carve = [&](size_t bytes) -> void* {
    void* p = (char*)d_ws + off;
    off += (bytes + 255) & ~(size_t)255;
    return p;
  };
  float*    xp    = (float*)   carve((size_t)N * 256 * 4);   // post-GEMM features
  float*    hbuf  = (float*)   carve((size_t)N * 256 * 4);   // agg / layer output
  _Float16* h16   = (_Float16*)carve((size_t)N * 256 * 2);   // f16 activations
  _Float16* W16p  = (_Float16*)carve((size_t)256 * 256 * 2); // fragment-packed weights
  float*    sbuf  = (float*)   carve((size_t)N * 4 * 4);
  float*    dbuf  = (float*)   carve((size_t)N * 4 * 4);
  float*    emax  = (float*)   carve((size_t)N * 4 * 4);
  float*    denom = (float*)   carve((size_t)N * 4 * 4);
  float*    alpha = (float*)   carve((size_t)ET * 4 * 4);
  (void)ws_size; (void)n_in;

  const int BT = 256;

  for (int L = 0; L < 3; ++L) {
    const int K = Cin[L], CO = Cout[L], H = Hh[L];
    const float* hin = (L == 0) ? x : hbuf;

    // 1) activations f32 -> f16; weights f32 -> fragment-packed f16
    cvt_f16_kernel<<<cdiv(N * K, BT), BT, 0, stream>>>(hin, h16, N * K);
    pack_w_kernel<<<cdiv(K * CO, BT), BT, 0, stream>>>(Wl[L], W16p, K, CO);

    // 2) xp = h16 @ W (WMMA, f32 accumulate)
    dim3 ggrid(cdiv(CO, 64), cdiv(N, 16));
    gemm_f16_wmma_kernel<<<ggrid, 128, 0, stream>>>(h16, W16p, xp, N, K, CO);

    // 3) attention logits per node
    sd_kernel<<<cdiv(N * H, BT), BT, 0, stream>>>(xp, asrc[L], adst[L], sbuf, dbuf, N, H);

    // 4) init segment buffers
    fill_f32_kernel<<<cdiv(N * H, BT), BT, 0, stream>>>(emax, -INFINITY, N * H);
    fill_f32_kernel<<<cdiv(N * H, BT), BT, 0, stream>>>(denom, 0.f, N * H);
    fill_f32_kernel<<<cdiv(N * CO, BT), BT, 0, stream>>>(hbuf, 0.f, N * CO);

    // 5) softmax over incoming edges (3 passes)
    edge_max_kernel<<<cdiv(ET * H, BT), BT, 0, stream>>>(ei, E, ET, sbuf, dbuf, emax, H);
    edge_ee_kernel <<<cdiv(ET * H, BT), BT, 0, stream>>>(ei, E, ET, sbuf, dbuf, emax, alpha, denom, H);
    alpha_div_kernel<<<cdiv(ET * H, BT), BT, 0, stream>>>(ei, E, ET, alpha, denom, H);

    // 6) message aggregation (dominant memory phase)
    int tpe = CO >> 2;
    long long totalT = (long long)ET * tpe;
    edge_agg_kernel<<<(int)((totalT + BT - 1) / BT), BT, 0, stream>>>(ei, E, ET, alpha, xp, hbuf, H, CO);

    // 7) bias + ReLU (in place; hbuf becomes next layer input)
    bias_relu_kernel<<<cdiv(N * CO, BT), BT, 0, stream>>>(hbuf, bl[L], N * CO, CO);
  }

  // final FC: [N,64] @ [64,10] + b
  fc_kernel<<<cdiv(N, BT), BT, 0, stream>>>(hbuf, Wfc, bfc, out, N);
}
